// MultiHeadSelfAttention_31001073943224
// MI455X (gfx1250) — compile-verified, matrix operand order checked by reading, not run on hardware
//
#include <hip/hip_runtime.h>
#include <math.h>

// ---------------------------------------------------------------------------
// Multi-head self-attention, MI455X (gfx1250), bf16 WMMA pipeline with
// TDM-staged (tensor_load_to_lds) B panels and register double-buffering.
//   B=8, N=1024, C=768, H=12, D=64, SCALE=1/8
// ---------------------------------------------------------------------------

#define BATCH     8
#define SEQ       1024
#define DIM       768
#define NHEADS    12
#define HDIM      64
#define SCALE_F   0.125f
#define LOG2E_F   1.44269504088896f
#define NK        (DIM / 32)          // 24 k-steps

typedef __attribute__((ext_vector_type(8)))  __bf16 v8bf;
typedef __attribute__((ext_vector_type(16))) __bf16 v16bf;
typedef __attribute__((ext_vector_type(8)))  float  v8f;
typedef __attribute__((ext_vector_type(4)))  unsigned int v4u;
typedef __attribute__((ext_vector_type(4)))  int v4i;
typedef __attribute__((ext_vector_type(8)))  int v8i;

// Tensor Data Mover availability (device pass only; host pass parses fallback)
#if defined(__has_builtin)
#if __has_builtin(__builtin_amdgcn_tensor_load_to_lds) && \
    __has_builtin(__builtin_amdgcn_s_wait_tensorcnt)
#define USE_TDM 1
#endif
#endif
#ifndef USE_TDM
#define USE_TDM 0
#endif

// Load a 16-element bf16 fragment: two contiguous 8-element (16B) chunks at
// p and p+16.  Matches the CDNA5 16-bit A/B fragment layout: per lane the
// K-chunks are {base..base+7} and {base+16..base+23}.
__device__ __forceinline__ v16bf ldfrag(const __bf16* p) {
  v8bf lo = *(const v8bf*)p;
  v8bf hi = *(const v8bf*)(p + 16);
  return __builtin_shufflevector(lo, hi, 0,1,2,3,4,5,6,7,8,9,10,11,12,13,14,15);
}

__device__ __forceinline__ v8f wmma_bf16(v16bf a, v16bf b, v8f c) {
  return __builtin_amdgcn_wmma_f32_16x16x32_bf16(
      false, a, false, b, (short)0, c, false, false);
}

#if USE_TDM
// Issue a TDM load of a 64-row x 32-element bf16 tile (row stride DIM elems)
// from global memory into LDS at lds_addr.  D# per CDNA5 ISA ch.7/8:
//   group0: count=1 | lds_addr | global_addr(57b) | type=2
//   group1: data_size=1(2B), tensor_dim0=768, tensor_dim1=64,
//           tile_dim0=32, tile_dim1=64, tensor_dim0_stride=768
__device__ __forceinline__ void tdm_load_tile(unsigned int lds_addr,
                                              const __bf16* gptr) {
  unsigned long long ga = (unsigned long long)(uintptr_t)gptr;
  v4u g0;
  g0[0] = 1u;                                     // count=1, user descriptor
  g0[1] = lds_addr;                               // LDS byte address
  g0[2] = (unsigned int)ga;                       // global_addr[31:0]
  g0[3] = (unsigned int)((ga >> 32) & 0x01FFFFFFu) | 0x80000000u; // [56:32]|type=2
  v8i g1;
  g1[0] = (int)(1u << 16);                        // data_size = 1 (2 bytes)
  g1[1] = (int)(768u << 16);                      // tensor_dim0[15:0] = 768
  g1[2] = (int)(64u << 16);                       // dim0 hi=0 | tensor_dim1 = 64
  g1[3] = (int)(32u << 16);                       // dim1 hi=0 | tile_dim0 = 32
  g1[4] = 64;                                     // tile_dim1=64, tile_dim2=0
  g1[5] = 768;                                    // tensor_dim0_stride[31:0]
  g1[6] = 0;                                      // stride hi / dim1_stride
  g1[7] = 0;
  v4i z4 = {0, 0, 0, 0};
#if defined(__clang_major__) && (__clang_major__ >= 23)
  v8i z8 = {0, 0, 0, 0, 0, 0, 0, 0};
  __builtin_amdgcn_tensor_load_to_lds(g0, g1, z4, z4, z8, 0);
#else
  __builtin_amdgcn_tensor_load_to_lds(g0, g1, z4, z4, 0);
#endif
}
#endif

// ---------------------------------------------------------------------------
// f32 -> bf16 elementwise convert
// ---------------------------------------------------------------------------
__global__ void cvt_f32_bf16_kernel(const float* __restrict__ src,
                                    __bf16* __restrict__ dst, int n) {
  int i = blockIdx.x * blockDim.x + threadIdx.x;
  if (i < n) dst[i] = (__bf16)src[i];
}

// ---------------------------------------------------------------------------
// GEMM core: one wave computes a 32x64 tile; the 64-col B panel (shared by
// all 8 waves of the block) is TDM-staged through double-buffered LDS.
// A fragments register double-buffered; all prefetches unconditional (one
// step of overread stays inside the workspace by construction).
// ---------------------------------------------------------------------------
struct GemmOut { v8f acc[2][4]; };

__device__ __forceinline__ void gemm_tile_32x64(
    const __bf16* __restrict__ arow0,   // A row (mtile*32+lm), pre-offset half*8
    const __bf16* __restrict__ wpanel,  // W + (g*64)*DIM  (64-row B panel)
    __bf16* btile0, __bf16* btile1,     // 2 x 4KB LDS buffers
    int lane, GemmOut& out) {
  const int half = lane >> 4, lm = lane & 15;
  const __bf16* arow1 = arow0 + 16 * DIM;

#pragma unroll
  for (int r = 0; r < 2; ++r)
#pragma unroll
    for (int t = 0; t < 4; ++t) out.acc[r][t] = (v8f){};

#if USE_TDM
  // scalar wave-0 predicate: TDM ignores EXEC, so gate with a scalar branch
  const bool tdmw = (__builtin_amdgcn_readfirstlane(threadIdx.x >> 5) == 0);
  const unsigned int lds0 = (unsigned int)(uintptr_t)btile0;
  const unsigned int lds1 = (unsigned int)(uintptr_t)btile1;
  if (tdmw) {
    tdm_load_tile(lds0, wpanel);          // k-step 0
    tdm_load_tile(lds1, wpanel + 32);     // k-step 1
  }
  v16bf a0n = ldfrag(arow0), a1n = ldfrag(arow1);
#pragma unroll 2
  for (int kk = 0; kk < NK; ++kk) {
    if (tdmw) __builtin_amdgcn_s_wait_tensorcnt(1);  // btile[kk&1] landed
    __syncthreads();                      // ... and visible to all waves
    const v16bf a0 = a0n, a1 = a1n;
    a0n = ldfrag(arow0 + (kk + 1) * 32);  // next A frags in flight during WMMAs
    a1n = ldfrag(arow1 + (kk + 1) * 32);
    const __bf16* bb = (kk & 1) ? btile1 : btile0;
#pragma unroll
    for (int t = 0; t < 4; ++t) {
      v16bf bf = ldfrag(bb + (t * 16 + lm) * 32 + half * 8);   // ds_load_b128
      out.acc[0][t] = wmma_bf16(a0, bf, out.acc[0][t]);
      out.acc[1][t] = wmma_bf16(a1, bf, out.acc[1][t]);
    }
    __syncthreads();                      // block done reading btile[kk&1]
    if (tdmw)                             // refill freed buffer for kk+2
      tdm_load_tile((kk & 1) ? lds1 : lds0, wpanel + (kk + 2) * 32);
  }
#else
  // Fallback: direct global B fragments, register double-buffered A
  (void)btile0; (void)btile1;
  const __bf16* brow = wpanel + (size_t)lm * DIM + half * 8;
  v16bf a0n = ldfrag(arow0), a1n = ldfrag(arow1);
  for (int kk = 0; kk < NK; ++kk) {
    const int k = kk * 32;
    const v16bf a0 = a0n, a1 = a1n;
    a0n = ldfrag(arow0 + k + 32);
    a1n = ldfrag(arow1 + k + 32);
#pragma unroll
    for (int t = 0; t < 4; ++t) {
      v16bf bf = ldfrag(brow + (size_t)t * 16 * DIM + k);
      out.acc[0][t] = wmma_bf16(a0, bf, out.acc[0][t]);
      out.acc[1][t] = wmma_bf16(a1, bf, out.acc[1][t]);
    }
  }
#endif
}

// ---------------------------------------------------------------------------
// QKV GEMM:  X(8192x768) @ W^T -> Q,K (B,H,N,D) ; V^T (B,H,D,N).
// Block = 8 waves = 256 rows x 64 cols (one head's columns).
// ---------------------------------------------------------------------------
__global__ __launch_bounds__(256) void qkv_gemm_kernel(
    const __bf16* __restrict__ X, const __bf16* __restrict__ W,
    const float* __restrict__ bias,
    __bf16* __restrict__ Q, __bf16* __restrict__ K, __bf16* __restrict__ V) {
  __shared__ __attribute__((aligned(16))) __bf16 btile[2][64 * 32];
  const int lane = threadIdx.x & 31, w = threadIdx.x >> 5;
  const int half = lane >> 4, lm = lane & 15;
  const int mtile = blockIdx.x * 8 + w;   // 0..255 (32-row tiles over B*N)
  const int g = blockIdx.y;               // 0..35

  const __bf16* arow0 = X + (size_t)(mtile * 32 + lm) * DIM + half * 8;
  const __bf16* wpanel = W + (size_t)(g * 64) * DIM;

  GemmOut o;
  gemm_tile_32x64(arow0, wpanel, btile[0], btile[1], lane, o);

  const int sec = g / 12;     // 0=Q 1=K 2=V
  const int h   = g % 12;
#pragma unroll
  for (int r = 0; r < 2; ++r) {
#pragma unroll
    for (int t = 0; t < 4; ++t) {
      const int d = t * 16 + lm;
      const float bv = bias[sec * DIM + h * HDIM + d];
#pragma unroll
      for (int v = 0; v < 8; ++v) {
        const int m  = mtile * 32 + r * 16 + v + 8 * half;
        const int bb = m >> 10, nq = m & 1023;
        const size_t bh = (size_t)bb * NHEADS + h;
        const float val = o.acc[r][t][v] + bv;
        if (sec == 0)      Q[(bh * SEQ + nq) * HDIM + d] = (__bf16)(val * SCALE_F);
        else if (sec == 1) K[(bh * SEQ + nq) * HDIM + d] = (__bf16)val;
        else               V[(bh * HDIM + d) * SEQ + nq] = (__bf16)val;   // V^T
      }
    }
  }
}

// ---------------------------------------------------------------------------
// Flash attention step for one 32-key chunk.  Consumes K fragments from
// `cur`, unconditionally prefetches the next chunk's fragments into `nxt`
// (one chunk of overread lands in the V^T region of the workspace).
// ---------------------------------------------------------------------------
__device__ __forceinline__ void attn_step(
    int kb, const __bf16* __restrict__ Kp, const __bf16* __restrict__ Vp,
    __bf16* pb, int half, int lm,
    v16bf aq0, v16bf aq1, v16bf (&cur)[4], v16bf (&nxt)[4],
    float (&mrun)[8], float (&lrun)[8], v8f (&o)[4]) {
  const __bf16* kn = Kp + (size_t)(kb + 32) * HDIM;
  nxt[0] = ldfrag(kn);        nxt[1] = ldfrag(kn + 32);
  nxt[2] = ldfrag(kn + 1024); nxt[3] = ldfrag(kn + 1024 + 32);

  // --- scores: two 16x16 tiles, f32 acc ---------------------------------
  v8f s0 = {}, s1 = {};
  s0 = wmma_bf16(aq0, cur[0], s0);
  s0 = wmma_bf16(aq1, cur[1], s0);
  s1 = wmma_bf16(aq0, cur[2], s1);
  s1 = wmma_bf16(aq1, cur[3], s1);

  // --- online softmax stats (row = v + 8*half, over 16 lanes) -----------
  float p0[8], p1[8], al[8];
#pragma unroll
  for (int v = 0; v < 8; ++v) {
    float mx = fmaxf(s0[v], s1[v]);
#pragma unroll
    for (int d = 1; d < 16; d <<= 1) mx = fmaxf(mx, __shfl_xor(mx, d, 32));
    const float mnew = fmaxf(mrun[v], mx);
    al[v] = exp2f((mrun[v] - mnew) * LOG2E_F);
    p0[v] = exp2f((s0[v] - mnew) * LOG2E_F);
    p1[v] = exp2f((s1[v] - mnew) * LOG2E_F);
    mrun[v] = mnew;
    float r = p0[v] + p1[v];
#pragma unroll
    for (int d = 1; d < 16; d <<= 1) r += __shfl_xor(r, d, 32);
    lrun[v] = lrun[v] * al[v] + r;
  }
#pragma unroll
  for (int t = 0; t < 4; ++t)
#pragma unroll
    for (int v = 0; v < 8; ++v) o[t][v] *= al[v];

  // --- P tile: C/D layout -> LDS row-major -> A fragment ----------------
#pragma unroll
  for (int v = 0; v < 8; ++v) {
    pb[(v + 8 * half) * 32 + lm]      = (__bf16)p0[v];
    pb[(v + 8 * half) * 32 + 16 + lm] = (__bf16)p1[v];
  }
  asm volatile("s_wait_dscnt 0" ::: "memory");
  const v16bf pa = ldfrag(pb + lm * 32 + half * 8);

  // --- O += P(16x32) @ V(32x64): 4 column tiles of V^T ------------------
#pragma unroll
  for (int t = 0; t < 4; ++t) {
    const v16bf bv = ldfrag(Vp + (size_t)(t * 16 + lm) * SEQ + kb + half * 8);
    o[t] = wmma_bf16(pa, bv, o[t]);
  }
}

__global__ __launch_bounds__(128) void attn_kernel(
    const __bf16* __restrict__ Q, const __bf16* __restrict__ K,
    const __bf16* __restrict__ V, __bf16* __restrict__ O) {
  __shared__ __attribute__((aligned(16))) __bf16 pbuf[4][16 * 32];

  const int lane = threadIdx.x & 31, w = threadIdx.x >> 5;
  const int half = lane >> 4, lm = lane & 15;
  const int tile = blockIdx.x * 4 + w;    // 0 .. B*H*64-1
  const int qt = tile & 63;
  const int hb = tile >> 6;
  const int h = hb % NHEADS, b = hb / NHEADS;
  const size_t bh = (size_t)b * NHEADS + h;

  const __bf16* Qp = Q + (bh * SEQ + qt * 16) * HDIM;
  const __bf16* Kp = K + bh * SEQ * HDIM + (size_t)lm * HDIM + half * 8;
  const __bf16* Vp = V + bh * HDIM * SEQ;   // [D][N]
  __bf16* pb = pbuf[w];

  const __bf16* qrow = Qp + lm * HDIM + half * 8;
  const v16bf aq0 = ldfrag(qrow);        // d 0..31
  const v16bf aq1 = ldfrag(qrow + 32);   // d 32..63

  float mrun[8], lrun[8];
  v8f o[4] = {};
#pragma unroll
  for (int v = 0; v < 8; ++v) { mrun[v] = -3.0e38f; lrun[v] = 0.0f; }

  v16bf kfA[4], kfB[4];
  kfA[0] = ldfrag(Kp);           kfA[1] = ldfrag(Kp + 32);
  kfA[2] = ldfrag(Kp + 1024);    kfA[3] = ldfrag(Kp + 1024 + 32);

  for (int kb = 0; kb < SEQ; kb += 64) {   // explicit two-buffer ping-pong
    attn_step(kb,      Kp, Vp, pb, half, lm, aq0, aq1, kfA, kfB, mrun, lrun, o);
    attn_step(kb + 32, Kp, Vp, pb, half, lm, aq0, aq1, kfB, kfA, mrun, lrun, o);
  }

  // --- normalize and store to (B, N, C) bf16 for the proj GEMM ----------
#pragma unroll
  for (int v = 0; v < 8; ++v) {
    const float inv = 1.0f / lrun[v];
    const size_t row = (size_t)b * SEQ + qt * 16 + v + 8 * half;
#pragma unroll
    for (int t = 0; t < 4; ++t)
      O[row * DIM + h * HDIM + t * 16 + lm] = (__bf16)(o[t][v] * inv);
  }
}

// ---------------------------------------------------------------------------
// Proj GEMM: A(8192x768) @ W^T(768x768) + b -> out f32
// ---------------------------------------------------------------------------
__global__ __launch_bounds__(256) void proj_gemm_kernel(
    const __bf16* __restrict__ A, const __bf16* __restrict__ W,
    const float* __restrict__ bias, float* __restrict__ out) {
  __shared__ __attribute__((aligned(16))) __bf16 btile[2][64 * 32];
  const int lane = threadIdx.x & 31, w = threadIdx.x >> 5;
  const int half = lane >> 4, lm = lane & 15;
  const int mtile = blockIdx.x * 8 + w;   // 0..255
  const int g = blockIdx.y;               // 0..11

  const __bf16* arow0 = A + (size_t)(mtile * 32 + lm) * DIM + half * 8;
  const __bf16* wpanel = W + (size_t)(g * 64) * DIM;

  GemmOut o;
  gemm_tile_32x64(arow0, wpanel, btile[0], btile[1], lane, o);

#pragma unroll
  for (int r = 0; r < 2; ++r) {
#pragma unroll
    for (int t = 0; t < 4; ++t) {
      const int col = g * 64 + t * 16 + lm;
      const float bv = bias[col];
#pragma unroll
      for (int v = 0; v < 8; ++v) {
        const int m = mtile * 32 + r * 16 + v + 8 * half;
        out[(size_t)m * DIM + col] = o.acc[r][t][v] + bv;
      }
    }
  }
}

// ---------------------------------------------------------------------------
// Workspace layout (ordered so every one-step prefetch overread stays inside
// the allocation):  Xb | Wq | Wp | Ab | Qb | Kb | Vb
// ---------------------------------------------------------------------------
extern "C" void kernel_launch(void* const* d_in, const int* in_sizes, int n_in,
                              void* d_out, int out_size, void* d_ws, size_t ws_size,
                              hipStream_t stream) {
  const float* x      = (const float*)d_in[0];
  const float* qkv_w  = (const float*)d_in[1];
  const float* qkv_b  = (const float*)d_in[2];
  const float* proj_w = (const float*)d_in[3];
  const float* proj_b = (const float*)d_in[4];
  float* out = (float*)d_out;

  const size_t nX  = (size_t)BATCH * SEQ * DIM;           // 6 291 456
  const size_t nWq = (size_t)3 * DIM * DIM;               // 1 769 472
  const size_t nWp = (size_t)DIM * DIM;                   //   589 824
  const size_t nHd = (size_t)BATCH * NHEADS * SEQ * HDIM; // 6 291 456

  __bf16* Xb = (__bf16*)d_ws;
  __bf16* Wq = Xb + nX;
  __bf16* Wp = Wq + nWq;
  __bf16* Ab = Wp + nWp;   // attention output, (B,N,C) bf16
  __bf16* Qb = Ab + nX;
  __bf16* Kb = Qb + nHd;
  __bf16* Vb = Kb + nHd;

  cvt_f32_bf16_kernel<<<(int)((nX  + 255) / 256), 256, 0, stream>>>(x,      Xb, (int)nX);
  cvt_f32_bf16_kernel<<<(int)((nWq + 255) / 256), 256, 0, stream>>>(qkv_w,  Wq, (int)nWq);
  cvt_f32_bf16_kernel<<<(int)((nWp + 255) / 256), 256, 0, stream>>>(proj_w, Wp, (int)nWp);

  qkv_gemm_kernel<<<dim3(32, 36), 256, 0, stream>>>(Xb, Wq, qkv_b, Qb, Kb, Vb);
  attn_kernel<<<1536, 128, 0, stream>>>(Qb, Kb, Vb, Ab);
  proj_gemm_kernel<<<dim3(32, 12), 256, 0, stream>>>(Ab, Wp, proj_b, out);
}
